// MultiHeadSelfAttention_86139864088727
// MI455X (gfx1250) — matrix operand order repaired and checked by reading, not compiled, not run on hardware
//
#include <hip/hip_runtime.h>

// ---------------------------------------------------------------------------
// Causal multi-head self-attention for MI455X (gfx1250, wave32, WMMA).
// All GEMMs run through v_wmma_f32_16x16x32_bf16 (bf16 inputs, f32 accum).
// Shared operand tiles are staged into LDS with GLOBAL_LOAD_ASYNC_TO_LDS_B128
// (ASYNCcnt / s_wait_asynccnt), the CDNA5 async data-movement path.
//
// Shapes: B=4, T=2048, C=1024, H=16, hd=64.  M = B*T = 8192.
// Workspace layout (72 MiB total):
//   [ 0,16)MiB  x_bf   (bf16 copy of x; reused as y_bf after attention)
//   [16,22)MiB  wqT    (w_qkv transposed: 3072 x 1024, bf16)
//   [22,24)MiB  woT    (w_out transposed: 1024 x 1024, bf16)
//   [24,40)MiB  q_buf  ([B*H][T][64] bf16)
//   [40,56)MiB  k_buf  ([B*H][T][64] bf16)
//   [56,72)MiB  vT_buf ([B*H][64][T] bf16)  -- transposed for P@V B-frags
// ---------------------------------------------------------------------------

#define D_MODEL 1024
#define NHEAD   16
#define HD      64
#define BATCH   4
#define SEQ     2048
#define MROWS   (BATCH * SEQ) /* 8192 */

typedef __attribute__((ext_vector_type(16))) __bf16 v16bf;
typedef __attribute__((ext_vector_type(8)))  float  v8f;

union Frag {
  v16bf v;
  uint4 u[2];
};

__device__ __forceinline__ unsigned short f2bf(float f) {
  union { float f; unsigned int u; } a;
  a.f = f;
  unsigned int u = a.u;
  unsigned int r = (u + 0x7FFFu + ((u >> 16) & 1u)) >> 16; // RNE
  return (unsigned short)r;
}

__device__ __forceinline__ v8f wmma_bf16(Frag a, Frag b, v8f c) {
  // D = A(16x32) * B(32x16) + C(16x16), f32 accumulate
  return __builtin_amdgcn_wmma_f32_16x16x32_bf16(
      false, a.v, false, b.v, (short)0, c, false, false);
}

// Async copy of one 16-byte chunk per lane: global -> LDS, ASYNCcnt-tracked.
__device__ __forceinline__ void async_b128(unsigned lds_byte_addr,
                                           const void* gaddr) {
  asm volatile("global_load_async_to_lds_b128 %0, %1, off"
               :: "v"(lds_byte_addr), "v"(gaddr)
               : "memory");
}
__device__ __forceinline__ void wait_async0() {
  asm volatile("s_wait_asynccnt 0" ::: "memory");
}

// Truncate a generic pointer to the 32-bit LDS byte offset (aperture rule:
// flat LDS address low 32 bits are the LDS offset).
__device__ __forceinline__ unsigned lds_off(const void* p) {
  return (unsigned)(size_t)p;
}

// Row reductions across the 16 lanes of a half-wave (C-layout rows live on
// lanes [0,16) or [16,32); xor masks 1,2,4,8 never cross the half boundary).
__device__ __forceinline__ float rmax16(float v) {
  v = fmaxf(v, __shfl_xor(v, 1, 32));
  v = fmaxf(v, __shfl_xor(v, 2, 32));
  v = fmaxf(v, __shfl_xor(v, 4, 32));
  v = fmaxf(v, __shfl_xor(v, 8, 32));
  return v;
}
__device__ __forceinline__ float rsum16(float v) {
  v += __shfl_xor(v, 1, 32);
  v += __shfl_xor(v, 2, 32);
  v += __shfl_xor(v, 4, 32);
  v += __shfl_xor(v, 8, 32);
  return v;
}

// ---------------------------------------------------------------------------
// Conversion kernels
// ---------------------------------------------------------------------------
__global__ void cvt_bf16_kernel(const float* __restrict__ in,
                                unsigned short* __restrict__ out, int n) {
  int i = blockIdx.x * blockDim.x + threadIdx.x;
  if (i < n) out[i] = f2bf(in[i]);
}

// in: R x C (row-major f32)  ->  out: C x R (row-major bf16)
__global__ void cvt_transpose_bf16_kernel(const float* __restrict__ in,
                                          unsigned short* __restrict__ out,
                                          int R, int C) {
  int i = blockIdx.x * blockDim.x + threadIdx.x; // linear over output
  if (i < R * C) {
    int c = i / R;
    int r = i - c * R;
    out[i] = f2bf(in[(size_t)r * C + c]);
  }
}

// ---------------------------------------------------------------------------
// QKV projection: (8192 x 1024) @ (1024 x 3072) + bias, scatter into
// q[BH][T][64], k[BH][T][64], vT[BH][64][T] as bf16.
// Block = 4 waves stacked in M (128 x 64 block tile); each wave owns a
// 32 x 64 tile (8 accumulators, 8 WMMAs per K-step of 32).  The 64x32
// weight tile is async-staged into LDS once per K-step and shared.
// ---------------------------------------------------------------------------
__global__ __launch_bounds__(128) void qkv_gemm_kernel(
    const unsigned short* __restrict__ xb,   // M x 1024 bf16
    const unsigned short* __restrict__ wqT,  // 3072 x 1024 bf16 (N^T x K)
    const float* __restrict__ bias,          // 3072
    unsigned short* __restrict__ qb,
    unsigned short* __restrict__ kb,
    unsigned short* __restrict__ vtb) {
  __shared__ unsigned short Bsh[64 * 32]; // N-block x K-step tile, 4 KB

  const int tid    = threadIdx.x;
  const int wave   = tid >> 5;
  const int lane   = tid & 31;
  const int laneHi = lane >> 4;
  const int lmod   = lane & 15;

  const int NB = 3 * D_MODEL / 64; // 48 N-blocks
  const int mb = blockIdx.x / NB;
  const int nb = blockIdx.x - mb * NB;
  const int m0 = mb * 128 + wave * 32;
  const int n0 = nb * 64;

  const unsigned short* ar0 = xb + (size_t)(m0 + lmod) * D_MODEL;
  const unsigned short* ar1 = xb + (size_t)(m0 + 16 + lmod) * D_MODEL;

  v8f acc[2][4];
  {
    v8f z = {};
#pragma unroll
    for (int mm = 0; mm < 2; ++mm)
#pragma unroll
      for (int c = 0; c < 4; ++c) acc[mm][c] = z;
  }

  for (int k0 = 0; k0 < D_MODEL; k0 += 32) {
    __syncthreads(); // previous-iteration consumers done with Bsh
    // stage wqT[n0..n0+63][k0..k0+31] -> Bsh (row-major 64 x 32)
#pragma unroll
    for (int i = 0; i < 2; ++i) {
      const int ch = tid + i * 128;          // 256 chunks of 8 halfs
      const int r  = ch >> 2;                // n-local row
      const int c8 = (ch & 3) * 8;           // k-local col (halfs)
      async_b128(lds_off(&Bsh[r * 32 + c8]),
                 wqT + (size_t)(n0 + r) * D_MODEL + k0 + c8);
    }
    wait_async0();
    __syncthreads();

    Frag a0, a1;
    a0.u[0] = *(const uint4*)(ar0 + k0 + 8 * laneHi);
    a0.u[1] = *(const uint4*)(ar0 + k0 + 16 + 8 * laneHi);
    a1.u[0] = *(const uint4*)(ar1 + k0 + 8 * laneHi);
    a1.u[1] = *(const uint4*)(ar1 + k0 + 16 + 8 * laneHi);
#pragma unroll
    for (int c = 0; c < 4; ++c) {
      Frag b;
      const unsigned short* bp = &Bsh[(16 * c + lmod) * 32];
      b.u[0] = *(const uint4*)(bp + 8 * laneHi);
      b.u[1] = *(const uint4*)(bp + 16 + 8 * laneHi);
      acc[0][c] = wmma_bf16(a0, b, acc[0][c]);
      acc[1][c] = wmma_bf16(a1, b, acc[1][c]);
    }
  }

#pragma unroll
  for (int c = 0; c < 4; ++c) {
    const int n  = n0 + 16 * c + lmod;
    const float bv = bias[n];
    const int which = n >> 10;  // 0:q 1:k 2:v
    const int nn = n & 1023;
    const int h  = nn >> 6;
    const int d  = nn & 63;
#pragma unroll
    for (int mm = 0; mm < 2; ++mm) {
#pragma unroll
      for (int j = 0; j < 8; ++j) {
        const int m  = m0 + 16 * mm + j + 8 * laneHi;
        const int bb = m >> 11;
        const int t  = m & 2047;
        const int bh = bb * NHEAD + h;
        const unsigned short val = f2bf(acc[mm][c][j] + bv);
        if (which == 0)      qb [((size_t)bh * SEQ + t) * HD + d] = val;
        else if (which == 1) kb [((size_t)bh * SEQ + t) * HD + d] = val;
        else                 vtb[((size_t)bh * HD + d) * SEQ + t] = val;
      }
    }
  }
}

// ---------------------------------------------------------------------------
// Flash attention: block = 4 waves = 4 consecutive 16-query tiles of the SAME
// head (128 tiles/head, divisible by 4).  Key blocks of 32: K (32x64) and
// vT (64x32) tiles are async-staged into LDS once per block and shared by
// all 4 waves.  Waves past their causal bound skip compute but keep staging.
// ---------------------------------------------------------------------------
__global__ __launch_bounds__(128) void attn_kernel(
    const unsigned short* __restrict__ qb,
    const unsigned short* __restrict__ kb,
    const unsigned short* __restrict__ vtb,
    unsigned short* __restrict__ yb) { // M x 1024 bf16 out
  __shared__ unsigned short Ksh[32 * 64];     // 4 KB
  __shared__ unsigned short Vsh[64 * 32];     // 4 KB
  __shared__ unsigned short Psh[4][16 * 32];  // 4 KB (per-wave P transpose)

  const int tid    = threadIdx.x;
  const int wave   = tid >> 5;
  const int lane   = tid & 31;
  const int laneHi = lane >> 4;
  const int lmod   = lane & 15;

  const int tile0 = blockIdx.x * 4;        // wave0's tile
  const int bh  = tile0 >> 7;              // SEQ/16 = 128 tiles per head
  const int tq0 = (tile0 & 127) << 4;      // wave0's q0
  const int q0  = tq0 + wave * 16;

  const unsigned short* qbase = qb  + (size_t)bh * SEQ * HD;
  const unsigned short* kbase = kb  + (size_t)bh * SEQ * HD;
  const unsigned short* vbase = vtb + (size_t)bh * HD * SEQ;

  // Q fragments (16x64 = two K=32 A-frags), resident for the whole loop.
  Frag qf0, qf1;
  {
    const unsigned short* qr = qbase + (size_t)(q0 + lmod) * HD;
    qf0.u[0] = *(const uint4*)(qr + 8 * laneHi);
    qf0.u[1] = *(const uint4*)(qr + 16 + 8 * laneHi);
    qf1.u[0] = *(const uint4*)(qr + 32 + 8 * laneHi);
    qf1.u[1] = *(const uint4*)(qr + 48 + 8 * laneHi);
  }

  float m_i[8], l_i[8];
  v8f acc[4];
#pragma unroll
  for (int j = 0; j < 8; ++j) { m_i[j] = -1e30f; l_i[j] = 0.0f; }
  {
    v8f z = {};
#pragma unroll
    for (int c = 0; c < 4; ++c) acc[c] = z;
  }

  unsigned short* pl = &Psh[wave][0];
  const float sc = 0.125f;                 // 1/sqrt(64)
  const int nblk_own = (q0 + 47) >> 5;     // blocks this wave needs
  const int nblk_max = (tq0 + 95) >> 5;    // blocks the last wave needs

  for (int blk = 0; blk < nblk_max; ++blk) {
    const int j0 = blk << 5;

    __syncthreads(); // previous-iteration consumers done with Ksh/Vsh
    // stage K rows [j0, j0+32) x 64 (contiguous 4 KB in k_buf)
#pragma unroll
    for (int i = 0; i < 2; ++i) {
      const int ch = tid + i * 128; // 256 chunks of 8 halfs
      async_b128(lds_off(Ksh) + ch * 16,
                 kbase + (size_t)j0 * HD + ch * 8);
    }
    // stage vT rows d=0..63, cols [j0, j0+32)  (row stride SEQ in vT_buf)
#pragma unroll
    for (int i = 0; i < 2; ++i) {
      const int ch = tid + i * 128;
      const int r  = ch >> 2;
      const int c8 = (ch & 3) * 8;
      async_b128(lds_off(Vsh) + ch * 16,
                 vbase + (size_t)r * SEQ + j0 + c8);
    }
    wait_async0();
    __syncthreads();

    if (blk < nblk_own) { // uniform per wave -> EXEC stays all-ones
      // ---- scores: S(16x32) = Q(16x64) @ K^T ----
      v8f s0 = {}, s1 = {};
      {
        const unsigned short* kr0 = &Ksh[lmod * 64];
        const unsigned short* kr1 = &Ksh[(16 + lmod) * 64];
        Frag b;
        b.u[0] = *(const uint4*)(kr0 + 8 * laneHi);
        b.u[1] = *(const uint4*)(kr0 + 16 + 8 * laneHi);
        s0 = wmma_bf16(qf0, b, s0);
        b.u[0] = *(const uint4*)(kr0 + 32 + 8 * laneHi);
        b.u[1] = *(const uint4*)(kr0 + 48 + 8 * laneHi);
        s0 = wmma_bf16(qf1, b, s0);
        b.u[0] = *(const uint4*)(kr1 + 8 * laneHi);
        b.u[1] = *(const uint4*)(kr1 + 16 + 8 * laneHi);
        s1 = wmma_bf16(qf0, b, s1);
        b.u[0] = *(const uint4*)(kr1 + 32 + 8 * laneHi);
        b.u[1] = *(const uint4*)(kr1 + 48 + 8 * laneHi);
        s1 = wmma_bf16(qf1, b, s1);
      }

      // ---- online softmax per row ----
#pragma unroll
      for (int j = 0; j < 8; ++j) {
        const int qi  = q0 + j + 8 * laneHi;
        const int kj0 = j0 + lmod;
        const int kj1 = j0 + 16 + lmod;
        float a0 = (kj0 > qi) ? -1e30f : s0[j] * sc;
        float a1 = (kj1 > qi) ? -1e30f : s1[j] * sc;
        const float mx   = rmax16(fmaxf(a0, a1));
        const float mnew = fmaxf(m_i[j], mx);
        const float resc = __expf(m_i[j] - mnew);
        const float p0   = __expf(a0 - mnew);
        const float p1   = __expf(a1 - mnew);
        l_i[j] = l_i[j] * resc + rsum16(p0 + p1);
        m_i[j] = mnew;
#pragma unroll
        for (int c = 0; c < 4; ++c) acc[c][j] *= resc;
        const int row = j + 8 * laneHi;
        pl[row * 32 + lmod]      = f2bf(p0);
        pl[row * 32 + 16 + lmod] = f2bf(p1);
      }

      // ---- read P back in A layout (in-wave LDS ops are ordered) ----
      Frag pf;
      pf.u[0] = *(const uint4*)(pl + lmod * 32 + 8 * laneHi);
      pf.u[1] = *(const uint4*)(pl + lmod * 32 + 16 + 8 * laneHi);

      // ---- O(16x64) += P(16x32) @ V(32x64) via staged vT ----
#pragma unroll
      for (int c = 0; c < 4; ++c) {
        const unsigned short* vr = &Vsh[(16 * c + lmod) * 32];
        Frag bv;
        bv.u[0] = *(const uint4*)(vr + 8 * laneHi);
        bv.u[1] = *(const uint4*)(vr + 16 + 8 * laneHi);
        acc[c] = wmma_bf16(pf, bv, acc[c]);
      }
    }
  }

  // ---- epilogue: O /= l, write bf16 y in (B,T,C) layout ----
  const int bb = bh >> 4; // batch
  const int h  = bh & 15; // head
#pragma unroll
  for (int j = 0; j < 8; ++j) {
    const float inv = 1.0f / l_i[j];
    const int t = q0 + j + 8 * laneHi;
    const size_t rowbase = ((size_t)bb * SEQ + t) * D_MODEL + h * HD;
#pragma unroll
    for (int c = 0; c < 4; ++c) {
      yb[rowbase + 16 * c + lmod] = f2bf(acc[c][j] * inv);
    }
  }
}

// ---------------------------------------------------------------------------
// Output projection: (8192 x 1024) @ (1024 x 1024) + bias -> f32 d_out.
// Same blocking as qkv_gemm (128x64 block, 32x64 per wave, staged weights).
// ---------------------------------------------------------------------------
__global__ __launch_bounds__(128) void out_gemm_kernel(
    const unsigned short* __restrict__ yb,   // M x 1024 bf16
    const unsigned short* __restrict__ woT,  // 1024 x 1024 bf16 (N^T x K)
    const float* __restrict__ bias,          // 1024
    float* __restrict__ out) {               // M x 1024 f32
  __shared__ unsigned short Bsh[64 * 32];

  const int tid    = threadIdx.x;
  const int wave   = tid >> 5;
  const int lane   = tid & 31;
  const int laneHi = lane >> 4;
  const int lmod   = lane & 15;

  const int NB = D_MODEL / 64; // 16
  const int mb = blockIdx.x / NB;
  const int nb = blockIdx.x - mb * NB;
  const int m0 = mb * 128 + wave * 32;
  const int n0 = nb * 64;

  const unsigned short* ar0 = yb + (size_t)(m0 + lmod) * D_MODEL;
  const unsigned short* ar1 = yb + (size_t)(m0 + 16 + lmod) * D_MODEL;

  v8f acc[2][4];
  {
    v8f z = {};
#pragma unroll
    for (int mm = 0; mm < 2; ++mm)
#pragma unroll
      for (int c = 0; c < 4; ++c) acc[mm][c] = z;
  }

  for (int k0 = 0; k0 < D_MODEL; k0 += 32) {
    __syncthreads();
#pragma unroll
    for (int i = 0; i < 2; ++i) {
      const int ch = tid + i * 128;
      const int r  = ch >> 2;
      const int c8 = (ch & 3) * 8;
      async_b128(lds_off(&Bsh[r * 32 + c8]),
                 woT + (size_t)(n0 + r) * D_MODEL + k0 + c8);
    }
    wait_async0();
    __syncthreads();

    Frag a0, a1;
    a0.u[0] = *(const uint4*)(ar0 + k0 + 8 * laneHi);
    a0.u[1] = *(const uint4*)(ar0 + k0 + 16 + 8 * laneHi);
    a1.u[0] = *(const uint4*)(ar1 + k0 + 8 * laneHi);
    a1.u[1] = *(const uint4*)(ar1 + k0 + 16 + 8 * laneHi);
#pragma unroll
    for (int c = 0; c < 4; ++c) {
      Frag b;
      const unsigned short* bp = &Bsh[(16 * c + lmod) * 32];
      b.u[0] = *(const uint4*)(bp + 8 * laneHi);
      b.u[1] = *(const uint4*)(bp + 16 + 8 * laneHi);
      acc[0][c] = wmma_bf16(a0, b, acc[0][c]);
      acc[1][c] = wmma_bf16(a1, b, acc[1][c]);
    }
  }

#pragma unroll
  for (int c = 0; c < 4; ++c) {
    const int n = n0 + 16 * c + lmod;
    const float bv = bias[n];
#pragma unroll
    for (int mm = 0; mm < 2; ++mm) {
#pragma unroll
      for (int j = 0; j < 8; ++j) {
        const int m = m0 + 16 * mm + j + 8 * laneHi;
        out[(size_t)m * D_MODEL + n] = acc[mm][c][j] + bv;
      }
    }
  }
}

// ---------------------------------------------------------------------------
extern "C" void kernel_launch(void* const* d_in, const int* in_sizes, int n_in,
                              void* d_out, int out_size, void* d_ws, size_t ws_size,
                              hipStream_t stream) {
  const float* x     = (const float*)d_in[0]; // (4, 2048, 1024)
  const float* w_qkv = (const float*)d_in[1]; // (1024, 3072)
  const float* b_qkv = (const float*)d_in[2]; // (3072,)
  const float* w_out = (const float*)d_in[3]; // (1024, 1024)
  const float* b_out = (const float*)d_in[4]; // (1024,)
  float* out = (float*)d_out;
  char*  ws  = (char*)d_ws;

  unsigned short* x_bf  = (unsigned short*)(ws);                        // 16 MiB (reused as y)
  unsigned short* wqT   = (unsigned short*)(ws + ((size_t)16 << 20));   //  6 MiB
  unsigned short* woT   = (unsigned short*)(ws + ((size_t)22 << 20));   //  2 MiB
  unsigned short* q_buf = (unsigned short*)(ws + ((size_t)24 << 20));   // 16 MiB
  unsigned short* k_buf = (unsigned short*)(ws + ((size_t)40 << 20));   // 16 MiB
  unsigned short* vT_buf= (unsigned short*)(ws + ((size_t)56 << 20));   // 16 MiB

  // 1) precision conversion / weight transposes
  const int nx = MROWS * D_MODEL; // 8388608
  cvt_bf16_kernel<<<(nx + 255) / 256, 256, 0, stream>>>(x, x_bf, nx);
  const int nwq = D_MODEL * 3 * D_MODEL; // 3145728
  cvt_transpose_bf16_kernel<<<(nwq + 255) / 256, 256, 0, stream>>>(
      w_qkv, wqT, D_MODEL, 3 * D_MODEL);
  const int nwo = D_MODEL * D_MODEL; // 1048576
  cvt_transpose_bf16_kernel<<<(nwo + 255) / 256, 256, 0, stream>>>(
      w_out, woT, D_MODEL, D_MODEL);

  // 2) QKV projection: (8192/128) * (3072/64) = 64*48 = 3072 blocks
  qkv_gemm_kernel<<<3072, 128, 0, stream>>>(x_bf, wqT, b_qkv,
                                            q_buf, k_buf, vT_buf);

  // 3) flash attention: BH * (SEQ/16) / 4 = 2048 blocks
  attn_kernel<<<2048, 128, 0, stream>>>(q_buf, k_buf, vT_buf, x_bf);

  // 4) output projection: (8192/128) * (1024/64) = 64*16 = 1024 blocks
  out_gemm_kernel<<<1024, 128, 0, stream>>>(x_bf, woT, b_out, out);
}
